// TidalTransformer_36301063586542
// MI455X (gfx1250) — compile-verified, matrix operand order checked by reading, not run on hardware
//
#include <hip/hip_runtime.h>
#include <hip/hip_bf16.h>

// ---------------------------------------------------------------------------
// TidalTransformer on gfx1250 (CDNA5): f16 WMMA GEMMs + WMMA attention.
// GEMM uses double-buffered GLOBAL_LOAD_ASYNC_TO_LDS_B128 staging (ASYNCcnt)
// with v_wmma_f32_16x16x32_f16 compute.
// ---------------------------------------------------------------------------

#define D_    512
#define H_    16
#define HD_   32
#define FF_   2048
#define B_    8
#define SL_   433
#define OL_   144
#define CTX_  72
#define EPS_  1e-5f
#define NEG_  -1000000000.0f

typedef __attribute__((ext_vector_type(16))) _Float16 v16h;
typedef __attribute__((ext_vector_type(8)))  _Float16 v8h;
typedef __attribute__((ext_vector_type(8)))  float    v8f;

__device__ __forceinline__ v8f wmma_f16(v16h a, v16h b, v8f c) {
    // (neg_a, A, neg_b, B, c_mod, C, reuse_a, reuse_b)
    return __builtin_amdgcn_wmma_f32_16x16x32_f16(false, a, false, b,
                                                  (short)0, c, false, false);
}
__device__ __forceinline__ v16h cat8(v8h lo, v8h hi) {
    return __builtin_shufflevector(lo, hi, 0,1,2,3,4,5,6,7,8,9,10,11,12,13,14,15);
}
__device__ __forceinline__ float wave_sum(float v) {
#pragma unroll
    for (int m = 16; m >= 1; m >>= 1) v += __shfl_xor(v, m, 32);
    return v;
}
__device__ __forceinline__ float wave_max(float v) {
#pragma unroll
    for (int m = 16; m >= 1; m >>= 1) v = fmaxf(v, __shfl_xor(v, m, 32));
    return v;
}

// LDS byte address (AS3 offset) of a generic pointer to a __shared__ object.
__device__ __forceinline__ unsigned lds_addr_of(const void* p) {
    return (unsigned)(size_t)(__attribute__((address_space(3))) const void*)p;
}
// Async 16B global -> LDS copy (per-lane addresses), tracked by ASYNCcnt.
__device__ __forceinline__ void async_ld16(unsigned lds_off, const float* g) {
    unsigned long long ga = (unsigned long long)(size_t)g;
    asm volatile("global_load_async_to_lds_b128 %0, %1, off"
                 :: "v"(lds_off), "v"(ga) : "memory");
}

// ---------------------------------------------------------------------------
// GEMM: C[M,N] = A[M,K] * W[N,K]^T + bias[N], optional ReLU.
// 4 waves / block, 64x64 tile, K stepped by 32. Tiles staged as f32 via
// double-buffered async-to-LDS (8 async instructions per wave per stage,
// so s_wait_asynccnt 8 == "previous batch landed"). f32 -> f16 conversion
// is folded into the fragment load (VALU co-executes with XDL WMMA).
//
// A-fragment (16-bit 16x32): lane(0..15) row m, K in {0..7,16..23};
// lane(16..31) same row, K in {8..15,24..31}. B uses the identical pattern
// with lane = output column (row of W), so one loader serves both.
//
// NOTE: staging is unguarded on M edges; every A operand lives in d_ws with
// live buffers after it and N/K are always tile multiples, so OOB tile rows
// read harmless in-workspace data and are never stored (epilogue guards).
// ---------------------------------------------------------------------------
#define AFS_ 36   // LDS row stride in f32 (32 + 4 pad; rows stay 16B aligned)

__global__ __launch_bounds__(128) void gemm_bias_kernel(
    const float* __restrict__ A, int lda,
    const float* __restrict__ W, int ldw,
    const float* __restrict__ bias,
    float* __restrict__ C, int ldc,
    int M, int N, int K, int relu)
{
    __shared__ float Af[2][64][AFS_];
    __shared__ float Bf[2][64][AFS_];
    const int tid  = threadIdx.x;
    const int lane = tid & 31, wave = tid >> 5;
    const int m0 = blockIdx.y * 64, n0 = blockIdx.x * 64;
    const int half = lane >> 4, lm = lane & 15, kb = half * 8;

    v8f acc[4];
#pragma unroll
    for (int n = 0; n < 4; n++) acc[n] = (v8f){0.f,0.f,0.f,0.f,0.f,0.f,0.f,0.f};

    // per-thread staging coordinates: 4 x (row, 16B segment) per matrix
    int srow[4], scol[4];
#pragma unroll
    for (int p = 0; p < 4; p++) {
        int i = tid + p * 128;            // 0..511
        srow[p] = i >> 3;                 // 0..63
        scol[p] = (i & 7) * 4;            // 0,4,...,28 (f32)
    }
    const float* Abase = A + (size_t)m0 * lda;
    const float* Wbase = W + (size_t)n0 * ldw;

    auto stage = [&](int bsel, int k0) {
#pragma unroll
        for (int p = 0; p < 4; p++) {
            async_ld16(lds_addr_of(&Af[bsel][srow[p]][scol[p]]),
                       Abase + (size_t)srow[p] * lda + k0 + scol[p]);
            async_ld16(lds_addr_of(&Bf[bsel][srow[p]][scol[p]]),
                       Wbase + (size_t)srow[p] * ldw + k0 + scol[p]);
        }
    };
    auto cvt_frag = [&](const float* p) {
        v16h f;
#pragma unroll
        for (int j = 0; j < 8; j++) {
            f[j]     = (_Float16)p[j];
            f[j + 8] = (_Float16)p[16 + j];
        }
        return f;
    };

    const int nk = K >> 5;
    stage(0, 0);                                   // prefetch first tile
    for (int kt = 0; kt < nk; kt++) {
        const int cur = kt & 1;
        if (kt + 1 < nk) {
            stage(cur ^ 1, (kt + 1) << 5);         // prefetch next tile
            asm volatile("s_wait_asynccnt 8" ::: "memory");  // older batch done
        } else {
            asm volatile("s_wait_asynccnt 0" ::: "memory");
        }
        __syncthreads();                           // all waves' copies landed
        v16h af = cvt_frag(&Af[cur][wave * 16 + lm][kb]);
#pragma unroll
        for (int n = 0; n < 4; n++) {
            v16h bf = cvt_frag(&Bf[cur][n * 16 + lm][kb]);
            acc[n] = wmma_f16(af, bf, acc[n]);
        }
        __syncthreads();                           // done reading before reuse
    }

    // D layout: lane l, elem r -> row (r + 8*(l>>4)), col (l&15)
#pragma unroll
    for (int n = 0; n < 4; n++) {
#pragma unroll
        for (int r = 0; r < 8; r++) {
            int m = m0 + wave * 16 + r + half * 8;
            int c = n0 + n * 16 + lm;
            if (m < M && c < N) {
                float v = acc[n][r] + bias[c];
                if (relu) v = fmaxf(v, 0.f);
                C[(size_t)m * ldc + c] = v;
            }
        }
    }
}

// ---------------------------------------------------------------------------
// Attention: one wave per (b, h, q-tile of 16). Scores via WMMA (K=hd=32),
// softmax in LDS, P*V via WMMA over 32-wide K tiles.
// mode 0: no mask. mode 1: window mask (k <= q && k >= mstart) else NEG.
// ---------------------------------------------------------------------------
#define LKP_ 448   // max padded Lk (433 -> 448)

__global__ __launch_bounds__(32) void attn_kernel(
    const float* __restrict__ Q,  int ldq,
    const float* __restrict__ Kp, int ldk,
    const float* __restrict__ Vp, int ldv,
    float* __restrict__ O, int Lq, int Lk, int mode, int mstart)
{
    __shared__ float    s_sc[16][LKP_ + 4];
    __shared__ _Float16 s_pr[16][LKP_ + 8];
    __shared__ float    s_rs[16];

    const int lane = threadIdx.x;
    const int qt = blockIdx.x, h = blockIdx.y, b = blockIdx.z;
    const int q0 = qt * 16;
    const int half = lane >> 4, lm = lane & 15, kb = half * 8;
    const float scale = 0.17677669529663687f;   // 1/sqrt(32)

    // Q fragment (16 rows x 32 d), f32 -> f16 on the fly
    v16h aq;
    {
        int q = q0 + lm;
        const float* qp = (q < Lq) ? (Q + ((size_t)b * Lq + q) * ldq + h * HD_) : nullptr;
#pragma unroll
        for (int j = 0; j < 8; j++) {
            aq[j]     = qp ? (_Float16)qp[kb + j]      : (_Float16)0.f;
            aq[j + 8] = qp ? (_Float16)qp[kb + 16 + j] : (_Float16)0.f;
        }
    }

    // scores: S = Q K^T * scale (+ mask) into LDS
    const int nkt = (Lk + 15) >> 4;
    for (int kt = 0; kt < nkt; kt++) {
        int k0 = kt * 16;
        v16h bk;
        {
            int kc = k0 + lm;
            const float* kp = (kc < Lk) ? (Kp + ((size_t)b * Lk + kc) * ldk + h * HD_) : nullptr;
#pragma unroll
            for (int j = 0; j < 8; j++) {
                bk[j]     = kp ? (_Float16)kp[kb + j]      : (_Float16)0.f;
                bk[j + 8] = kp ? (_Float16)kp[kb + 16 + j] : (_Float16)0.f;
            }
        }
        v8f s = (v8f){0.f,0.f,0.f,0.f,0.f,0.f,0.f,0.f};
        s = wmma_f16(aq, bk, s);
#pragma unroll
        for (int r = 0; r < 8; r++) {
            int mrow = r + half * 8;
            int kc = k0 + lm;
            if (kc < Lk) {
                float sv = s[r] * scale;
                if (mode == 1) {
                    int qr = q0 + mrow;
                    if (!(kc <= qr && kc >= mstart)) sv = NEG_;
                }
                s_sc[mrow][kc] = sv;
            }
        }
    }

    // softmax rows (wave-cooperative); probs stored f16, zero-padded to Lkp
    const int Lkp = (Lk + 31) & ~31;
    for (int r = 0; r < 16; r++) {
        float mx = -3.0e38f;
        for (int c = lane; c < Lk; c += 32) mx = fmaxf(mx, s_sc[r][c]);
        mx = wave_max(mx);
        float sm = 0.f;
        for (int c = lane; c < Lk; c += 32) {
            float e = __expf(s_sc[r][c] - mx);
            s_pr[r][c] = (_Float16)e;
            sm += e;
        }
        sm = wave_sum(sm);
        if (lane == 0) s_rs[r] = sm;
        for (int c = Lk + lane; c < Lkp; c += 32) s_pr[r][c] = (_Float16)0.f;
    }

    // O = P V ; two 16-wide d halves; normalize by row sum in epilogue
    for (int dh = 0; dh < 2; dh++) {
        v8f acc = (v8f){0.f,0.f,0.f,0.f,0.f,0.f,0.f,0.f};
        int d = dh * 16 + lm;
        for (int ks = 0; ks < Lkp; ks += 32) {
            v8h lo = *(const v8h*)&s_pr[lm][ks + kb];
            v8h hi = *(const v8h*)&s_pr[lm][ks + kb + 16];
            v16h ap = cat8(lo, hi);
            v16h bv;
#pragma unroll
            for (int j = 0; j < 8; j++) {
                int k1 = ks + kb + j, k2 = ks + kb + 16 + j;
                bv[j]     = (k1 < Lk) ? (_Float16)Vp[((size_t)b * Lk + k1) * ldv + h * HD_ + d]
                                      : (_Float16)0.f;
                bv[j + 8] = (k2 < Lk) ? (_Float16)Vp[((size_t)b * Lk + k2) * ldv + h * HD_ + d]
                                      : (_Float16)0.f;
            }
            acc = wmma_f16(ap, bv, acc);
        }
#pragma unroll
        for (int r = 0; r < 8; r++) {
            int mrow = r + half * 8;
            int q = q0 + mrow;
            if (q < Lq)
                O[((size_t)b * Lq + q) * D_ + h * HD_ + dh * 16 + lm] = acc[r] / s_rs[mrow];
        }
    }
}

// ---------------------------------------------------------------------------
// LayerNorm (+ optional residual): y = LN(x + r) * g + b. One wave per row.
// ---------------------------------------------------------------------------
__global__ __launch_bounds__(128) void ln_kernel(
    const float* __restrict__ x, const float* __restrict__ res,
    const float* __restrict__ g, const float* __restrict__ bb,
    float* __restrict__ y, int rows)
{
    int lane = threadIdx.x & 31;
    int row  = blockIdx.x * 4 + (threadIdx.x >> 5);
    if (row >= rows) return;
    const float* xr = x + (size_t)row * D_;
    const float* rr = res ? res + (size_t)row * D_ : nullptr;
    float v[16];
    float s = 0.f;
#pragma unroll
    for (int i = 0; i < 16; i++) {
        int c = lane + i * 32;
        float t = xr[c] + (rr ? rr[c] : 0.f);
        v[i] = t; s += t;
    }
    s = wave_sum(s);
    float mean = s * (1.f / 512.f);
    float q = 0.f;
#pragma unroll
    for (int i = 0; i < 16; i++) { float d = v[i] - mean; q += d * d; }
    q = wave_sum(q);
    float rstd = rsqrtf(q * (1.f / 512.f) + EPS_);
#pragma unroll
    for (int i = 0; i < 16; i++) {
        int c = lane + i * 32;
        y[(size_t)row * D_ + c] = (v[i] - mean) * rstd * g[c] + bb[c];
    }
}

// input projection (K=1) + sinusoidal positional encoding
__global__ void inproj_kernel(const float* __restrict__ src,
                              const float* __restrict__ w,
                              const float* __restrict__ bias,
                              float* __restrict__ x)
{
    int idx = blockIdx.x * 256 + threadIdx.x;
    if (idx >= B_ * SL_ * D_) return;
    int d = idx & (D_ - 1);
    int l = (idx / D_) % SL_;
    float div = __expf((float)(d & ~1) * (-9.210340371976184f / 512.f));
    float pe  = (d & 1) ? __cosf((float)l * div) : __sinf((float)l * div);
    x[idx] = src[idx / D_] * w[d] + bias[d] + pe;
}

__global__ void zero_kernel(float* __restrict__ p, int n)
{
    int i = blockIdx.x * 256 + threadIdx.x;
    if (i < n) p[i] = 0.f;
}

// buf' = concat(buf[:,1:,:], hnorm[:,-1,:])
__global__ void shift_kernel(const float* __restrict__ ob,
                             const float* __restrict__ hn,
                             float* __restrict__ nb)
{
    int idx = blockIdx.x * 256 + threadIdx.x;
    if (idx >= B_ * CTX_ * D_) return;
    int d = idx & (D_ - 1);
    int j = (idx / D_) % CTX_;
    int b = idx / (D_ * CTX_);
    nb[idx] = (j < CTX_ - 1) ? ob[idx + D_]
                             : hn[((size_t)b * CTX_ + (CTX_ - 1)) * D_ + d];
}

// out[b, step] = dot(hnorm[b, -1, :], w) + b0
__global__ __launch_bounds__(32) void outproj_kernel(
    const float* __restrict__ hn, const float* __restrict__ w,
    const float* __restrict__ bb, float* __restrict__ out, int step)
{
    int b = blockIdx.x, lane = threadIdx.x;
    const float* row = hn + ((size_t)b * CTX_ + (CTX_ - 1)) * D_;
    float s = 0.f;
    for (int i = lane; i < D_; i += 32) s += row[i] * w[i];
    s = wave_sum(s);
    if (lane == 0) out[b * OL_ + step] = s + bb[0];
}

// ---------------------------------------------------------------------------
// Host orchestration (graph-capture safe: only kernel launches on `stream`).
// Input flattening assumption: depth-first, dict-insertion order of
// setup_inputs(): src; in_proj_w, in_proj_b, out_proj_w, out_proj_b;
// enc[0..2]{sa{in_w,in_b,out_w,out_b}, ln1{g,b}, ln2{g,b}, w1,b1,w2,b2};
// enc_norm{g,b}; dec{sa{..}, ca{..}, ln1,ln2,ln3, w1,b1,w2,b2}; dec_norm{g,b}.
// ---------------------------------------------------------------------------
extern "C" void kernel_launch(void* const* d_in, const int* in_sizes, int n_in,
                              void* d_out, int out_size, void* d_ws, size_t ws_size,
                              hipStream_t stream)
{
    (void)in_sizes; (void)n_in; (void)out_size; (void)ws_size;
    int t = 0;
    auto nx = [&]() { return (const float*)d_in[t++]; };

    const float* src        = nx();
    const float* in_proj_w  = nx();
    const float* in_proj_b  = nx();
    const float* out_proj_w = nx();
    const float* out_proj_b = nx();

    struct MHA { const float *in_w, *in_b, *out_w, *out_b; };
    struct LNP { const float *g, *b; };
    struct ENC { MHA sa; LNP ln1, ln2; const float *w1,*b1,*w2,*b2; } enc[3];
    for (int i = 0; i < 3; i++) {
        enc[i].sa  = {nx(), nx(), nx(), nx()};
        enc[i].ln1 = {nx(), nx()};
        enc[i].ln2 = {nx(), nx()};
        enc[i].w1 = nx(); enc[i].b1 = nx(); enc[i].w2 = nx(); enc[i].b2 = nx();
    }
    LNP enc_norm = {nx(), nx()};
    struct DEC { MHA sa, ca; LNP ln1, ln2, ln3; const float *w1,*b1,*w2,*b2; } dec;
    dec.sa  = {nx(), nx(), nx(), nx()};
    dec.ca  = {nx(), nx(), nx(), nx()};
    dec.ln1 = {nx(), nx()}; dec.ln2 = {nx(), nx()}; dec.ln3 = {nx(), nx()};
    dec.w1 = nx(); dec.b1 = nx(); dec.w2 = nx(); dec.b2 = nx();
    LNP dec_norm = {nx(), nx()};

    // workspace layout (A operands of edge-M GEMMs must not be last: dq is last)
    float* ws = (float*)d_ws;
    size_t off = 0;
    auto alloc = [&](size_t n) { float* p = ws + off; off += (n + 63) & ~(size_t)63; return p; };
    const int ME = B_ * SL_;    // 3464 rows
    const int MD = B_ * CTX_;   // 576 rows
    float* xA    = alloc((size_t)ME * D_);
    float* xB    = alloc((size_t)ME * D_);
    float* big   = alloc((size_t)ME * FF_);   // qkv (ld 1536) / ffn hidden (ld 2048)
    float* attO  = alloc((size_t)ME * D_);
    float* prjO  = alloc((size_t)ME * D_);
    float* mem   = alloc((size_t)ME * D_);
    float* kc    = alloc((size_t)ME * D_);    // cross-attn K (precomputed)
    float* vc    = alloc((size_t)ME * D_);    // cross-attn V (precomputed)
    float* dbuf0 = alloc((size_t)MD * D_);
    float* dbuf1 = alloc((size_t)MD * D_);
    float* dbig  = alloc((size_t)MD * FF_);
    float* dat   = alloc((size_t)MD * D_);
    float* dt1   = alloc((size_t)MD * D_);
    float* dx2   = alloc((size_t)MD * D_);
    float* dx3   = alloc((size_t)MD * D_);
    float* dx4   = alloc((size_t)MD * D_);
    float* dnm   = alloc((size_t)MD * D_);
    float* dq    = alloc((size_t)MD * D_);

    auto gemm = [&](const float* A, int lda, const float* W, int ldw,
                    const float* bias, float* C, int ldc,
                    int M, int N, int K, int relu) {
        dim3 g((N + 63) / 64, (M + 63) / 64);
        gemm_bias_kernel<<<g, dim3(128), 0, stream>>>(A, lda, W, ldw, bias, C, ldc, M, N, K, relu);
    };
    auto ln = [&](const float* x, const float* r, LNP p, float* y, int rows) {
        ln_kernel<<<dim3((rows + 3) / 4), dim3(128), 0, stream>>>(x, r, p.g, p.b, y, rows);
    };
    auto attn = [&](const float* Q, int ldq, const float* Kp, int ldk,
                    const float* Vp, int ldv, float* O,
                    int Lq, int Lk, int mode, int mstart) {
        dim3 g((Lq + 15) / 16, H_, B_);
        attn_kernel<<<g, dim3(32), 0, stream>>>(Q, ldq, Kp, ldk, Vp, ldv, O, Lq, Lk, mode, mstart);
    };

    // ---------------- encoder ----------------
    inproj_kernel<<<dim3((ME * D_ + 255) / 256), dim3(256), 0, stream>>>(
        src, in_proj_w, in_proj_b, xA);
    for (int l = 0; l < 3; l++) {
        gemm(xA, D_, enc[l].sa.in_w, D_, enc[l].sa.in_b, big, 3 * D_, ME, 3 * D_, D_, 0);
        attn(big, 3 * D_, big + D_, 3 * D_, big + 2 * D_, 3 * D_, attO, SL_, SL_, 0, 0);
        gemm(attO, D_, enc[l].sa.out_w, D_, enc[l].sa.out_b, prjO, D_, ME, D_, D_, 0);
        ln(xA, prjO, enc[l].ln1, xB, ME);
        gemm(xB, D_, enc[l].w1, D_, enc[l].b1, big, FF_, ME, FF_, D_, 1);
        gemm(big, FF_, enc[l].w2, FF_, enc[l].b2, prjO, D_, ME, D_, FF_, 0);
        ln(xB, prjO, enc[l].ln2, xA, ME);
    }
    ln(xA, nullptr, enc_norm, mem, ME);

    // hoisted cross-attention K/V projections of mem (step-invariant)
    gemm(mem, D_, dec.ca.in_w + (size_t)D_ * D_,     D_, dec.ca.in_b + D_,     kc, D_, ME, D_, D_, 0);
    gemm(mem, D_, dec.ca.in_w + (size_t)2 * D_ * D_, D_, dec.ca.in_b + 2 * D_, vc, D_, ME, D_, D_, 0);

    // ---------------- decoder (144 sequential steps) ----------------
    zero_kernel<<<dim3((MD * D_ + 255) / 256), dim3(256), 0, stream>>>(dbuf0, MD * D_);
    for (int i = 0; i < OL_; i++) {
        float* bin  = (i & 1) ? dbuf1 : dbuf0;
        float* bout = (i & 1) ? dbuf0 : dbuf1;
        int L = (i + 1 < CTX_) ? (i + 1) : CTX_;
        int start = CTX_ - L;

        // self-attention (sliding-window causal mask)
        gemm(bin, D_, dec.sa.in_w, D_, dec.sa.in_b, dbig, 3 * D_, MD, 3 * D_, D_, 0);
        attn(dbig, 3 * D_, dbig + D_, 3 * D_, dbig + 2 * D_, 3 * D_, dat, CTX_, CTX_, 1, start);
        gemm(dat, D_, dec.sa.out_w, D_, dec.sa.out_b, dt1, D_, MD, D_, D_, 0);
        ln(bin, dt1, dec.ln1, dx2, MD);

        // cross-attention (K/V precomputed)
        gemm(dx2, D_, dec.ca.in_w, D_, dec.ca.in_b, dq, D_, MD, D_, D_, 0);
        attn(dq, D_, kc, D_, vc, D_, dat, CTX_, SL_, 0, 0);
        gemm(dat, D_, dec.ca.out_w, D_, dec.ca.out_b, dt1, D_, MD, D_, D_, 0);
        ln(dx2, dt1, dec.ln2, dx3, MD);

        // FFN
        gemm(dx3, D_, dec.w1, D_, dec.b1, dbig, FF_, MD, FF_, D_, 1);
        gemm(dbig, FF_, dec.w2, FF_, dec.b2, dt1, D_, MD, D_, FF_, 0);
        ln(dx3, dt1, dec.ln3, dx4, MD);

        // final norm, token output, buffer shift
        ln(dx4, nullptr, dec_norm, dnm, MD);
        outproj_kernel<<<dim3(B_), dim3(32), 0, stream>>>(
            dnm, out_proj_w, out_proj_b, (float*)d_out, i);
        shift_kernel<<<dim3((MD * D_ + 255) / 256), dim3(256), 0, stream>>>(bin, dnm, bout);
    }
}
